// Seq2SeqEncoder_1632087572960
// MI455X (gfx1250) — matrix operand order repaired and checked by reading, not compiled, not run on hardware
//
#include <hip/hip_runtime.h>
#include <hip/hip_bf16.h>

// ---------------------------------------------------------------------------
// Seq2Seq encoder: embedding + 2-layer LSTM (B=32, T=256, E=512, H=1024)
// bf16 WMMA (v_wmma_f32_16x16x32_bf16) for all matmuls, fp32 state/gates.
// ---------------------------------------------------------------------------

typedef __attribute__((ext_vector_type(16))) __bf16 v16bf;
typedef __attribute__((ext_vector_type(8)))  __bf16 v8bf;
typedef __attribute__((ext_vector_type(8)))  float  v8f;

#define Tn 256
#define Bn 32
#define En 512
#define Hn 1024
#define NG 4096            // 4*H gate width
#define REC_BLOCKS 32u     // persistent recurrent kernel blocks

// ---- WMMA helpers ----------------------------------------------------------

__device__ __forceinline__ v8f wmma_bf16(v16bf a, v16bf b, v8f c) {
  // 8 args: (neg_a, A, neg_b, B, c_mod, C, reuse_a, reuse_b)
  return __builtin_amdgcn_wmma_f32_16x16x32_bf16(false, a, false, b, (short)0, c,
                                                 false, false);
}

// A-matrix 16x32 bf16 fragment (ISA 7.12.2): row M = lane&15;
// lanes 0-15 hold K {0..7,16..23}, lanes 16-31 hold K {8..15,24..31}
// => two 16B chunks per lane at k + (lane>>4)*8 and +16 elements.
__device__ __forceinline__ v16bf load_fragA(const __bf16* __restrict__ base,
                                            int row0, int ld, int k) {
  int lane = threadIdx.x & 31;
  const __bf16* p = base + (size_t)(row0 + (lane & 15)) * ld + k + ((lane >> 4) << 3);
  union { v16bf v; v8bf h[2]; } u;
  u.h[0] = *(const v8bf*)(p);
  u.h[1] = *(const v8bf*)(p + 16);
  return u.v;
}

// B-matrix 32x16 bf16 fragment: column N = lane&15; same K order as A:
// lanes 0-15 hold K {0..7,16..23}, lanes 16-31 hold K {8..15,24..31}
// W stored (N,K) row-major => two 16B chunks per lane at k + (lane>>4)*8 and +16.
__device__ __forceinline__ v16bf load_fragB(const __bf16* __restrict__ base,
                                            int row0, int ld, int k) {
  int lane = threadIdx.x & 31;
  const __bf16* p = base + (size_t)(row0 + (lane & 15)) * ld + k + ((lane >> 4) << 3);
  union { v16bf v; v8bf h[2]; } u;
  u.h[0] = *(const v8bf*)(p);
  u.h[1] = *(const v8bf*)(p + 16);
  return u.v;
}

__device__ __forceinline__ float sigm(float x) { return 1.0f / (1.0f + __expf(-x)); }

// ---- grid-wide sense-reversal barrier (persistent recurrent kernel) --------

__device__ __forceinline__ void grid_sync(unsigned* cnt, unsigned* gen, unsigned nblk) {
  __syncthreads();
  if (threadIdx.x == 0) {
    __threadfence();
    unsigned g = __hip_atomic_load(gen, __ATOMIC_ACQUIRE, __HIP_MEMORY_SCOPE_AGENT);
    unsigned arrived = atomicAdd(cnt, 1u);
    if (arrived == nblk - 1u) {
      atomicExch(cnt, 0u);
      __threadfence();
      atomicAdd(gen, 1u);
    } else {
      while (__hip_atomic_load(gen, __ATOMIC_ACQUIRE, __HIP_MEMORY_SCOPE_AGENT) == g) {}
    }
    __threadfence();
  }
  __syncthreads();
}

// ---- small utility kernels -------------------------------------------------

__global__ void cvt_f32_bf16_kernel(const float* __restrict__ in,
                                    __bf16* __restrict__ out, int n) {
  int idx = blockIdx.x * blockDim.x + threadIdx.x;
  int stride = gridDim.x * blockDim.x;
  for (; idx < n; idx += stride) out[idx] = (__bf16)in[idx];
}

// X[(t*B+b)*E + e] = emb[source[b*T+t]*E + e]   (embedding gather + transpose)
__global__ void embed_kernel(const int* __restrict__ src,
                             const float* __restrict__ emb,
                             __bf16* __restrict__ X) {
  int idx = blockIdx.x * blockDim.x + threadIdx.x;
  const int total = Tn * Bn * En;
  int stride = gridDim.x * blockDim.x;
  for (; idx < total; idx += stride) {
    int e = idx & (En - 1);
    int r = idx >> 9;          // t*B + b
    int b = r & (Bn - 1);
    int t = r >> 5;
    int v = src[b * Tn + t];
    X[idx] = (__bf16)emb[(size_t)v * En + e];
  }
}

__global__ void init_state_kernel(unsigned* __restrict__ sync,
                                  __bf16* __restrict__ h_bf,
                                  float* __restrict__ c) {
  int idx = blockIdx.x * blockDim.x + threadIdx.x;
  if (idx < 2) sync[idx] = 0u;
  if (idx < Bn * Hn) {
    h_bf[idx] = (__bf16)0.0f;
    c[idx] = 0.0f;
  }
}

// ---- input-projection GEMM: C(Mtot,4096) = A(Mtot,K) @ W(4096,K)^T + bias --
// One wave computes a 32x32 output tile; 4 WMMAs per 32-deep K step.

__global__ void gemm_bf16_kernel(const __bf16* __restrict__ A,
                                 const __bf16* __restrict__ W,
                                 const float* __restrict__ bias,
                                 float* __restrict__ C, int Mtot, int K) {
  const int tilesN = NG / 32;  // 128
  int wid = blockIdx.x * (blockDim.x >> 5) + (threadIdx.x >> 5);
  int tM = wid / tilesN, tN = wid % tilesN;
  int m0 = tM * 32, n0 = tN * 32;
  if (m0 >= Mtot) return;
  int lane = threadIdx.x & 31;

  v8f acc00 = {0,0,0,0,0,0,0,0}, acc01 = {0,0,0,0,0,0,0,0};
  v8f acc10 = {0,0,0,0,0,0,0,0}, acc11 = {0,0,0,0,0,0,0,0};

  for (int k = 0; k < K; k += 32) {
    v16bf a0 = load_fragA(A, m0,      K, k);
    v16bf a1 = load_fragA(A, m0 + 16, K, k);
    v16bf b0 = load_fragB(W, n0,      K, k);
    v16bf b1 = load_fragB(W, n0 + 16, K, k);
    acc00 = wmma_bf16(a0, b0, acc00);
    acc01 = wmma_bf16(a0, b1, acc01);
    acc10 = wmma_bf16(a1, b0, acc10);
    acc11 = wmma_bf16(a1, b1, acc11);
  }

  int nA = n0 + (lane & 15), nB = nA + 16;
  float bA = bias[nA], bB = bias[nB];
  int mbase = m0 + ((lane >> 4) << 3);
#pragma unroll
  for (int r = 0; r < 8; ++r) {
    int mA = mbase + r, mB = mA + 16;
    C[(size_t)mA * NG + nA] = acc00[r] + bA;
    C[(size_t)mA * NG + nB] = acc01[r] + bB;
    C[(size_t)mB * NG + nA] = acc10[r] + bA;
    C[(size_t)mB * NG + nB] = acc11[r] + bB;
  }
}

// ---- persistent recurrent LSTM layer ---------------------------------------
// 32 blocks x 256 threads = 256 waves; wave w owns gate columns [16w, 16w+16).
// Per step: gates(32x4096) = G[t] + h @ Whh^T  (WMMA), grid sync,
// fused elementwise LSTM cell update (4 units/thread), grid sync.

__global__ void lstm_rec_kernel(const float* __restrict__ G,
                                const __bf16* __restrict__ Whh,
                                float* __restrict__ gs,
                                __bf16* __restrict__ h_bf,
                                float* __restrict__ c,
                                __bf16* __restrict__ y_bf,   // layer0: (T*B, H) bf16 or null
                                float* __restrict__ top,     // layer1: (B, T, H) f32 or null
                                float* __restrict__ hF,      // (B, H)
                                float* __restrict__ cF,      // (B, H)
                                unsigned* __restrict__ sync) {
  int wid = blockIdx.x * 8 + (threadIdx.x >> 5);   // 0..255
  int n0 = wid * 16;
  int lane = threadIdx.x & 31;
  int tid = blockIdx.x * blockDim.x + threadIdx.x; // 0..8191

  for (int t = 0; t < Tn; ++t) {
    // ---- recurrent matmul: 32 x 16 tile of h @ Whh^T, K = 1024 ----
    v8f acc0 = {0,0,0,0,0,0,0,0};
    v8f acc1 = {0,0,0,0,0,0,0,0};
    for (int k = 0; k < Hn; k += 32) {
      v16bf bf = load_fragB(Whh, n0, Hn, k);
      v16bf a0 = load_fragA(h_bf, 0,  Hn, k);
      v16bf a1 = load_fragA(h_bf, 16, Hn, k);
      acc0 = wmma_bf16(a0, bf, acc0);
      acc1 = wmma_bf16(a1, bf, acc1);
    }
    {
      int n = n0 + (lane & 15);
      int mb = (lane >> 4) << 3;
      const float* Gt = G + (size_t)t * Bn * NG;
#pragma unroll
      for (int r = 0; r < 8; ++r) {
        int bA = mb + r, bB = bA + 16;
        gs[(size_t)bA * NG + n] = acc0[r] + Gt[(size_t)bA * NG + n];
        gs[(size_t)bB * NG + n] = acc1[r] + Gt[(size_t)bB * NG + n];
      }
    }
    grid_sync(sync, sync + 1, REC_BLOCKS);

    // ---- fused LSTM cell update: 32*1024 units over 8192 threads ----
#pragma unroll
    for (int i = 0; i < 4; ++i) {
      int idx = tid * 4 + i;            // b*H + j
      int b = idx >> 10;
      int j = idx & (Hn - 1);
      const float* gb = gs + (size_t)b * NG;
      float gi = gb[j];
      float gf = gb[Hn + j];
      float gg = gb[2 * Hn + j];
      float go = gb[3 * Hn + j];
      float cn = sigm(gf) * c[idx] + sigm(gi) * tanhf(gg);
      float hn = sigm(go) * tanhf(cn);
      c[idx] = cn;
      h_bf[idx] = (__bf16)hn;
      if (y_bf) y_bf[((size_t)t * Bn + b) * Hn + j] = (__bf16)hn;
      if (top)  top[(size_t)b * Tn * Hn + (size_t)t * Hn + j] = hn;
      if (t == Tn - 1) { hF[idx] = hn; cF[idx] = cn; }
    }
    grid_sync(sync, sync + 1, REC_BLOCKS);
  }
}

// ---------------------------------------------------------------------------

extern "C" void kernel_launch(void* const* d_in, const int* in_sizes, int n_in,
                              void* d_out, int out_size, void* d_ws, size_t ws_size,
                              hipStream_t stream) {
  (void)in_sizes; (void)n_in; (void)out_size; (void)ws_size;

  const int*   src  = (const int*)d_in[0];
  const float* emb  = (const float*)d_in[1];
  const float* Wih0 = (const float*)d_in[2];
  const float* Whh0 = (const float*)d_in[3];
  const float* b0   = (const float*)d_in[4];
  const float* Wih1 = (const float*)d_in[5];
  const float* Whh1 = (const float*)d_in[6];
  const float* b1   = (const float*)d_in[7];

  float* out = (float*)d_out;
  float* top = out;                                   // (B, T, H)
  float* hF  = out + (size_t)Bn * Tn * Hn;            // (2, B, H)
  float* cF  = hF + 2 * (size_t)Bn * Hn;              // (2, B, H)

  // workspace carve-out (256B aligned chunks)
  char* w = (char*)d_ws;
  size_t off = 0;
  auto carve = [&](size_t bytes) -> void* {
    off = (off + 255) & ~(size_t)255;
    void* p = w + off;
    off += bytes;
    return p;
  };
  unsigned* sync    = (unsigned*)carve(256);
  __bf16* Wih0_bf   = (__bf16*)carve((size_t)NG * En * 2);
  __bf16* Whh0_bf   = (__bf16*)carve((size_t)NG * Hn * 2);
  __bf16* Wih1_bf   = (__bf16*)carve((size_t)NG * Hn * 2);
  __bf16* Whh1_bf   = (__bf16*)carve((size_t)NG * Hn * 2);
  __bf16* Xemb      = (__bf16*)carve((size_t)Tn * Bn * En * 2);
  __bf16* Y0        = (__bf16*)carve((size_t)Tn * Bn * Hn * 2);
  float*  Gbuf      = (float*)carve((size_t)Tn * Bn * NG * 4);  // shared L0/L1
  float*  gs        = (float*)carve((size_t)Bn * NG * 4);
  __bf16* h_bf      = (__bf16*)carve((size_t)Bn * Hn * 2);
  float*  cstate    = (float*)carve((size_t)Bn * Hn * 4);

  const int Mtot = Tn * Bn;  // 8192
  const int gemm_blocks = (Mtot / 32) * (NG / 32) / 8;  // 4096

  // weights -> bf16
  cvt_f32_bf16_kernel<<<2048, 256, 0, stream>>>(Wih0, Wih0_bf, NG * En);
  cvt_f32_bf16_kernel<<<2048, 256, 0, stream>>>(Whh0, Whh0_bf, NG * Hn);
  cvt_f32_bf16_kernel<<<2048, 256, 0, stream>>>(Wih1, Wih1_bf, NG * Hn);
  cvt_f32_bf16_kernel<<<2048, 256, 0, stream>>>(Whh1, Whh1_bf, NG * Hn);

  // embedding gather (+ (B,T)->(T,B) transpose), bf16
  embed_kernel<<<4096, 256, 0, stream>>>(src, emb, Xemb);

  // ---- layer 0 ----
  init_state_kernel<<<128, 256, 0, stream>>>(sync, h_bf, cstate);
  gemm_bf16_kernel<<<gemm_blocks, 256, 0, stream>>>(Xemb, Wih0_bf, b0, Gbuf, Mtot, En);
  lstm_rec_kernel<<<REC_BLOCKS, 256, 0, stream>>>(Gbuf, Whh0_bf, gs, h_bf, cstate,
                                                  Y0, nullptr, hF, cF, sync);

  // ---- layer 1 ----
  init_state_kernel<<<128, 256, 0, stream>>>(sync, h_bf, cstate);
  gemm_bf16_kernel<<<gemm_blocks, 256, 0, stream>>>(Y0, Wih1_bf, b1, Gbuf, Mtot, Hn);
  lstm_rec_kernel<<<REC_BLOCKS, 256, 0, stream>>>(Gbuf, Whh1_bf, gs, h_bf, cstate,
                                                  nullptr, top, hF + (size_t)Bn * Hn,
                                                  cF + (size_t)Bn * Hn, sync);
}
